// KitNET_30829275251300
// MI455X (gfx1250) — matrix operand order repaired and checked by reading, not compiled, not run on hardware
//
#include <hip/hip_runtime.h>

// KitNET (10 ensemble AEs + master) anomaly scorer for MI455X (gfx1250, wave32).
//
// Roofline: reads x once (460 MB) + writes 44 MB ~= 0.5 GB -> ~22 us at 23.3 TB/s.
// Compute (6.4 GFLOP) is trivial for WMMA, so the design minimizes per-wave
// instruction count: weights are pre-swizzled once into WMMA A-fragment layout
// (zero-padded) in d_ws, biases ride in the WMMA C operand, and all per-lane
// data movement is unconditional + vectorized (no exec-mask branches).

typedef _Float16 half_t;
typedef __attribute__((ext_vector_type(16))) _Float16 v16h;
typedef __attribute__((ext_vector_type(8)))  float    v8f;
typedef __attribute__((ext_vector_type(2)))  _Float16 h2v;

#define NFEAT 115
#define XREG  1856          // per-wave LDS floats: 16*115 data + 16 zero pad
#define WAVES 4
#define BLOCK_T (WAVES * 32)

constexpr int NAE = 11;
constexpr int DDIM[NAE] = {12, 10, 10, 8, 12, 10, 8, 8, 8, 29, 10};
constexpr int XSTART[10] = {0, 12, 22, 32, 40, 52, 62, 70, 78, 86};

constexpr int hdim(int d) { return (3 * d) / 4; }      // == int(d*0.75)
// WMMA tiles per AE: L1 (ceil(h/16)) + L2 (1) + L3 (ceil(h/16)) + L4 (ceil(d/16))
constexpr int tiles_in_ae(int d) {
  int h = hdim(d);
  return (h > 16 ? 2 : 1) * 2 + 1 + (d > 16 ? 2 : 1);
}
constexpr int tile_base(int ae) {
  int o = 0;
  for (int i = 0; i < ae; ++i) o += tiles_in_ae(DDIM[i]);
  return o;
}
constexpr int NT = tile_base(NAE);           // 47 tiles, 512 halfs each
constexpr int FRAG_HALFS = NT * 512;         // 24064 halfs = 48128 B (32B aligned)
constexpr int BIAS_FLOATS = NAE * 4 * 32;    // 4 padded bias vectors per AE

struct PtrPack { const float* p[88]; };      // 11 AEs x (W1,b1,W2,b2,W3,b3,W4,b4)

union U16 { unsigned u[8]; v16h h; };

__device__ inline v8f wmma_f16(v16h a, v16h b, v8f c) {
  return __builtin_amdgcn_wmma_f32_16x16x32_f16(false, a, false, b, (short)0, c,
                                                false, false);
}

__device__ inline unsigned pk16(float a, float b) {
  return __builtin_bit_cast(unsigned, __builtin_amdgcn_cvt_pkrtz(a, b));
}

__device__ inline void relu8(v8f& d) {
#pragma unroll
  for (int v = 0; v < 8; ++v) d[v] = fmaxf(d[v], 0.f);
}
__device__ inline void sig8(v8f& d) {
#pragma unroll
  for (int v = 0; v < 8; ++v)
    d[v] = __builtin_amdgcn_rcpf(1.f + __expf(-d[v]));
}

// B fragment (32x16 f16): column n = lane&15, K = (lane>>4)*16 + j.
// Reads 16 consecutive floats unconditionally; columns >= layer-K are
// multiplied by zero-padded A columns, so no masking is needed.
__device__ inline v16h load_bx(const float* __restrict__ p) {
  U16 u;
#pragma unroll
  for (int v = 0; v < 8; ++v) u.u[v] = pk16(p[2 * v], p[2 * v + 1]);
  return u.h;
}

// Repack D tiles (d0 rows 0-15, d1 rows 16-31; rows >= M are exact zeros)
// into the next layer's B fragment. Branchless: pk-convert, 1 shfl per dword,
// cndmask on half-select.
template <bool TWO>
__device__ inline v16h pack_b(v8f d0, v8f d1) {
  const int hlf = (threadIdx.x & 31) >> 4;
  unsigned h0[4], h1[4], o0[4], o1[4];
#pragma unroll
  for (int v = 0; v < 4; ++v) {
    h0[v] = pk16(d0[2 * v], d0[2 * v + 1]);
    h1[v] = TWO ? pk16(d1[2 * v], d1[2 * v + 1]) : 0u;
  }
#pragma unroll
  for (int v = 0; v < 4; ++v) {
    o0[v] = (unsigned)__shfl_xor((int)h0[v], 16);
    o1[v] = TWO ? (unsigned)__shfl_xor((int)h1[v], 16) : 0u;
  }
  U16 u;
#pragma unroll
  for (int v = 0; v < 4; ++v) {
    u.u[v]     = hlf ? o1[v] : h0[v];   // K = hlf*16 + [0..7]
    u.u[4 + v] = hlf ? h1[v] : o0[v];   // K = hlf*16 + [8..15]
  }
  return u.h;
}

// ---- AE core. fw -> this AE's fragment tiles (+lane*16 pre-added by caller is
// NOT done; we add here). bb -> this AE's 4x32 padded biases. Bias rides in C.
template <int DD, int HH, int HH2>
__device__ inline void ae_core(const half_t* __restrict__ fw,
                               const float* __restrict__ bb,
                               v16h bin, v8f& r0, v8f& r1) {
  const int lane = threadIdx.x & 31;
  const int hlf = lane >> 4;
  constexpr int L1T = (HH > 16) ? 2 : 1;
  constexpr int T2 = L1T;
  constexpr int T3 = T2 + 1;
  constexpr int T4 = T3 + L1T;
  const half_t* fl = fw + lane * 16;
  const v8f zz = {0.f, 0.f, 0.f, 0.f, 0.f, 0.f, 0.f, 0.f};

  v8f d0 = *(const v8f*)(bb + 0 * 32 + hlf * 8);
  d0 = wmma_f16(*(const v16h*)(fl + 0 * 512), bin, d0);
  v8f d1 = zz;
  if constexpr (HH > 16) {
    d1 = *(const v8f*)(bb + 0 * 32 + 16 + hlf * 8);
    d1 = wmma_f16(*(const v16h*)(fl + 1 * 512), bin, d1);
  }
  relu8(d0);
  if constexpr (HH > 16) relu8(d1);
  v16h b1 = pack_b<(HH > 16)>(d0, d1);

  v8f e0 = *(const v8f*)(bb + 1 * 32 + hlf * 8);   // HH2 <= 10: single tile
  e0 = wmma_f16(*(const v16h*)(fl + T2 * 512), b1, e0);
  relu8(e0);
  v16h b2 = pack_b<false>(e0, e0);

  v8f f0 = *(const v8f*)(bb + 2 * 32 + hlf * 8);
  f0 = wmma_f16(*(const v16h*)(fl + T3 * 512), b2, f0);
  v8f f1 = zz;
  if constexpr (HH > 16) {
    f1 = *(const v8f*)(bb + 2 * 32 + 16 + hlf * 8);
    f1 = wmma_f16(*(const v16h*)(fl + (T3 + 1) * 512), b2, f1);
  }
  relu8(f0);
  if constexpr (HH > 16) relu8(f1);
  v16h b3 = pack_b<(HH > 16)>(f0, f1);

  r0 = *(const v8f*)(bb + 3 * 32 + hlf * 8);
  r0 = wmma_f16(*(const v16h*)(fl + T4 * 512), b3, r0);
  if constexpr (DD > 16) {
    r1 = *(const v8f*)(bb + 3 * 32 + 16 + hlf * 8);
    r1 = wmma_f16(*(const v16h*)(fl + (T4 + 1) * 512), b3, r1);
  } else {
    r1 = zz;
  }
  sig8(r0);
  if constexpr (DD > 16) sig8(r1);
}

// Sum-of-squares of (x - recon) over one D tile; unconditional loads, cndmask.
template <int DD>
__device__ inline float resid_ss(const float* __restrict__ row, int hlf,
                                 v8f r, int m0) {
  float ss = 0.f;
#pragma unroll
  for (int v = 0; v < 8; ++v) {
    const int m = m0 + hlf * 8 + v;
    const float e = row[m] - r[v];     // over-read lands in pad / next row; masked
    const float em = (m < DD) ? e : 0.f;
    ss += em * em;
  }
  return ss;
}

template <int DD, int HH, int HH2>
__device__ inline float ae_ens(const half_t* __restrict__ fw,
                               const float* __restrict__ bb,
                               const float* __restrict__ xsw, int start) {
  const int lane = threadIdx.x & 31;
  const int nl = lane & 15, hlf = lane >> 4;
  const float* row = xsw + nl * NFEAT + start;
  v16h bin = load_bx(row + hlf * 16);
  v8f r0, r1;
  ae_core<DD, HH, HH2>(fw, bb, bin, r0, r1);
  float ss = resid_ss<DD>(row, hlf, r0, 0);
  if constexpr (DD > 16) ss += resid_ss<DD>(row, hlf, r1, 16);
  ss += __shfl_xor(ss, 16);
  return sqrtf(ss * (1.f / (float)DD));
}

// ---- One-shot setup: swizzle all weights into zero-padded A-fragment layout
// and pad biases to 32 floats. Tile order must match tile_base()/ae_core.
__global__ void kitnet_swizzle(PtrPack pp, half_t* __restrict__ frag,
                               float* __restrict__ bias) {
  int tile = 0, ai = 0;
  for (int ae = 0; ae < NAE; ++ae) {
    const int d = DDIM[ae], h = (3 * d) / 4, h2 = h / 2;
    const float* W[4] = {pp.p[ai + 0], pp.p[ai + 2], pp.p[ai + 4], pp.p[ai + 6]};
    const float* Bv[4] = {pp.p[ai + 1], pp.p[ai + 3], pp.p[ai + 5], pp.p[ai + 7]};
    const int Ms[4] = {h, h2, h, d};
    const int Ks[4] = {d, h, h2, h};
    for (int l = 0; l < 4; ++l) {
      for (int m0 = 0; m0 < Ms[l]; m0 += 16) {
        half_t* dst = frag + (size_t)tile * 512;
        for (int e = threadIdx.x; e < 512; e += blockDim.x) {
          const int lane = e >> 4, j = e & 15;
          const int m = (lane & 15) + m0;
          const int k = (lane >> 4) * 16 + j;
          half_t v = (half_t)0;
          if (m < Ms[l] && k < Ks[l]) v = (half_t)W[l][m * Ks[l] + k];
          dst[e] = v;
        }
        ++tile;
      }
      float* bdst = bias + (ae * 4 + l) * 32;
      for (int e = threadIdx.x; e < 32; e += blockDim.x)
        bdst[e] = (e < Ms[l]) ? Bv[l][e] : 0.f;
    }
    ai += 8;
  }
}

__global__ __launch_bounds__(BLOCK_T) void kitnet_main(
    const half_t* __restrict__ frag, const float* __restrict__ bias,
    const float* __restrict__ xg, float* __restrict__ out_an,
    float* __restrict__ out_en, int n) {
  __shared__ float s_x[WAVES * XREG];

  const int wave = threadIdx.x >> 5;
  const int lane = threadIdx.x & 31;
  const int nl = lane & 15, hlf = lane >> 4;
  const long long sampleBase = ((long long)blockIdx.x * WAVES + wave) * 16;
  const bool active = sampleBase < n;
  float* xsw = s_x + wave * XREG;

  if (active) {
    // 16 contiguous sample rows = 460 aligned float4s -> b128 load/store.
    const float4* src = (const float4*)xg;
    const long long b4 = sampleBase * NFEAT / 4;
    const long long max4 = (long long)n * NFEAT / 4;
    for (int i = lane; i < 460; i += 32) {
      long long gi = b4 + i;
      if (gi >= max4) gi = max4 - 1;          // tail clamp
      *(float4*)(xsw + 4 * i) = src[gi];
    }
    if (lane < 16) xsw[1840 + lane] = 0.f;    // zero the over-read pad
  }
  __syncthreads();
  if (!active) return;                        // wave-uniform; EXEC all-ones below

  float rm[10];
  rm[0] = ae_ens<12, 9, 4>(frag + tile_base(0) * 512, bias + 0 * 128, xsw, XSTART[0]);
  rm[1] = ae_ens<10, 7, 3>(frag + tile_base(1) * 512, bias + 1 * 128, xsw, XSTART[1]);
  rm[2] = ae_ens<10, 7, 3>(frag + tile_base(2) * 512, bias + 2 * 128, xsw, XSTART[2]);
  rm[3] = ae_ens<8, 6, 3>(frag + tile_base(3) * 512, bias + 3 * 128, xsw, XSTART[3]);
  rm[4] = ae_ens<12, 9, 4>(frag + tile_base(4) * 512, bias + 4 * 128, xsw, XSTART[4]);
  rm[5] = ae_ens<10, 7, 3>(frag + tile_base(5) * 512, bias + 5 * 128, xsw, XSTART[5]);
  rm[6] = ae_ens<8, 6, 3>(frag + tile_base(6) * 512, bias + 6 * 128, xsw, XSTART[6]);
  rm[7] = ae_ens<8, 6, 3>(frag + tile_base(7) * 512, bias + 7 * 128, xsw, XSTART[7]);
  rm[8] = ae_ens<8, 6, 3>(frag + tile_base(8) * 512, bias + 8 * 128, xsw, XSTART[8]);
  rm[9] = ae_ens<29, 21, 10>(frag + tile_base(9) * 512, bias + 9 * 128, xsw, XSTART[9]);

  // Master AE: B columns carry the 10 RMSEs in lanes 0-15 (K=0..9), zeros above.
  unsigned um[8];
  um[0] = pk16(rm[0], rm[1]);
  um[1] = pk16(rm[2], rm[3]);
  um[2] = pk16(rm[4], rm[5]);
  um[3] = pk16(rm[6], rm[7]);
  um[4] = pk16(rm[8], rm[9]);
  um[5] = um[6] = um[7] = 0u;
  U16 ub;
#pragma unroll
  for (int v = 0; v < 8; ++v) ub.u[v] = hlf ? 0u : um[v];

  v8f r0, r1;
  ae_core<10, 7, 3>(frag + tile_base(10) * 512, bias + 10 * 128, ub.h, r0, r1);

  float ss = 0.f;
#pragma unroll
  for (int v = 0; v < 8; ++v) {
    const int m = hlf * 8 + v;
    const float t = hlf ? (v < 2 ? rm[8 + v] : 0.f) : rm[v];
    const float e = (m < 10) ? (t - r0[v]) : 0.f;
    ss += e * e;
  }
  ss += __shfl_xor(ss, 16);
  const float score = sqrtf(ss * 0.1f);

  if (lane < 16) {
    const long long s = sampleBase + nl;
    if (s < n) {
      out_an[s] = score;
      float2* p2 = (float2*)(out_en + s * 10);
      p2[0] = make_float2(rm[0], rm[1]);
      p2[1] = make_float2(rm[2], rm[3]);
      p2[2] = make_float2(rm[4], rm[5]);
      p2[3] = make_float2(rm[6], rm[7]);
      p2[4] = make_float2(rm[8], rm[9]);
    }
  }
}

extern "C" void kernel_launch(void* const* d_in, const int* in_sizes, int n_in,
                              void* d_out, int out_size, void* d_ws, size_t ws_size,
                              hipStream_t stream) {
  (void)out_size; (void)ws_size;
  if (n_in < 1 + 88) return;
  const float* xg = (const float*)d_in[0];
  PtrPack pp;
  for (int i = 0; i < 88; ++i) pp.p[i] = (const float*)d_in[1 + i];

  half_t* frag = (half_t*)d_ws;                         // 48128 B
  float* bias = (float*)((char*)d_ws + FRAG_HALFS * 2); // 32B-aligned
  const int n = in_sizes[0] / NFEAT;
  float* out_an = (float*)d_out;
  float* out_en = out_an + n;

  hipLaunchKernelGGL(kitnet_swizzle, dim3(1), dim3(256), 0, stream, pp, frag, bias);

  const int wavesTotal = (n + 15) / 16;
  const int blocks = (wavesTotal + WAVES - 1) / WAVES;
  hipLaunchKernelGGL(kitnet_main, dim3(blocks), dim3(BLOCK_T), 0, stream,
                     frag, bias, xg, out_an, out_en, n);
}
